// RSSM_35639638622695
// MI455X (gfx1250) — matrix operand order repaired and checked by reading, not compiled, not run on hardware
//
#include <hip/hip_runtime.h>

// ---------------------------------------------------------------------------
// RSSM scan for MI455X (gfx1250). B=16 => one WMMA 16x16 tile covers batch.
// Weights converted once per launch to bf16, pre-swizzled into WMMA fragment
// layout and kept in workspace (~154MB -> fits 192MB L2, reused 64 steps).
// GEMM: 4 N-tiles per wave, K-extent templated so B-tile strides become
// immediate offsets; WGP-scope inline-asm prefetch warms the near cache.
// ---------------------------------------------------------------------------

#define B_ 16
#define T_ 64
#define DETER_ 4096
#define S_ 32
#define C_ 32
#define H_ 1024
#define ACT_ 6
#define E_ 1024
#define STOCH_ 1024
#define OUTD_ 7168  // DETER + STOCH + S*C + S*C

typedef __attribute__((ext_vector_type(16))) __bf16 v16bf;
typedef __attribute__((ext_vector_type(8)))  float  v8f;

__device__ __forceinline__ float sigm_(float x) { return 1.f / (1.f + __expf(-x)); }

__device__ __forceinline__ unsigned hash_u32(unsigned x) {
  x ^= x >> 16; x *= 0x7feb352dU; x ^= x >> 15; x *= 0x846ca68bU; x ^= x >> 16;
  return x;
}

// ---- A-fragment packing (16-bit A 16x32 layout, ISA 7.12.2) ---------------
// lane = m + 16*h, h = (kk%16)/8 ; slot = 8*(kk/16) + kk%8 ; kk = k%32
__device__ __forceinline__ size_t a_frag_off(int m, int k) {
  int kk = k & 31, ktile = k >> 5;
  int h = (kk & 15) >> 3, g = kk >> 4, kb = kk & 7;
  return (size_t)ktile * 512 + (size_t)(m + (h << 4)) * 16 + ((g << 3) + kb);
}

// ---- kernels ---------------------------------------------------------------

// Pack fp32 activation row-major (16 x Ksrc, leading dim ld) -> A fragments,
// zero-padding K up to Kpad (multiple of 32).
__global__ void pack_a(const float* __restrict__ src, long ld, int Ksrc, int Kpad,
                       __bf16* __restrict__ dst) {
  int i = blockIdx.x * blockDim.x + threadIdx.x;
  if (i >= 16 * Kpad) return;
  int m = i / Kpad, k = i - m * Kpad;
  float v = (k < Ksrc) ? src[(size_t)m * ld + k] : 0.f;
  dst[a_frag_off(m, k)] = (__bf16)v;
}

// Pack fp32 weight W[K x N] (row-major, k-major) -> B fragments, layout
// [ntile][ktile][lane][slot]; lane = n%16 + 16*((kk%16)/8), slot = 8*(kk/16) + kk%8.
__global__ void pack_b(const float* __restrict__ W, int Ksrc, int N, int Kb,
                       __bf16* __restrict__ dst) {
  size_t idx = (size_t)blockIdx.x * blockDim.x + threadIdx.x;
  size_t total = (size_t)(N >> 4) * Kb * 512;
  if (idx >= total) return;
  int slot = (int)(idx & 15);
  int lane = (int)((idx >> 4) & 31);
  size_t blk = idx >> 9;
  int ktile = (int)(blk % Kb);
  int ntile = (int)(blk / Kb);
  int k = ktile * 32 + ((slot >> 3) << 4) + ((lane >> 4) << 3) + (slot & 7);
  int n = (ntile << 4) + (lane & 15);
  float v = (k < Ksrc) ? W[(size_t)k * N + n] : 0.f;
  dst[idx] = (__bf16)v;
}

// C[16 x N] = A[16 x K] * B[K x N] (+ bias).
// Each wave computes NT=4 consecutive 16x16 N-tiles: av loaded once per
// k-step, 4 independent accumulator chains. KB compile-time => B-tile
// strides are immediate offsets (no per-iter 64-bit pointer math).
#define NT_ 4
template <int KB>
__global__ __launch_bounds__(128) void gemm16(
    const __bf16* __restrict__ Ap, const __bf16* __restrict__ Bp,
    const float* __restrict__ bias, float* __restrict__ C,
    int ntiles, long ldc) {
  int wave = threadIdx.x >> 5;
  int lane = threadIdx.x & 31;
  int tile0 = (blockIdx.x * (blockDim.x >> 5) + wave) * NT_;
  if (tile0 >= ntiles) return;  // wave-uniform
  constexpr size_t tstride = (size_t)KB * 512;  // elements per N-tile panel
  const __bf16* a = Ap + lane * 16;
  const __bf16* b = Bp + (size_t)tile0 * tstride + lane * 16;
  v8f acc[NT_] = {};
  for (int kb = 0; kb < KB; ++kb) {
#pragma unroll
    for (int i = 0; i < NT_; ++i) {
      const __bf16* bi = b + (size_t)i * tstride;
      v16bf av = *(const v16bf*)a;
      v16bf bv = *(const v16bf*)bi;
      // WGP-scope prefetch (scope 0): pulls the B stream into the near
      // cache; __builtin_prefetch only reaches SE/SYS scope (GL2-only).
      asm volatile("global_prefetch_b8 %0, off offset:4096" ::"v"(bi));
      acc[i] = __builtin_amdgcn_wmma_f32_16x16x32_bf16(
          false, av, false, bv, (short)0, acc[i], false, false);
    }
    a += 512;
    b += 512;
  }
  int m0 = (lane >> 4) * 8;
#pragma unroll
  for (int i = 0; i < NT_; ++i) {
    int n = (lane & 15) + (tile0 + i) * 16;
    float bs = bias ? bias[n] : 0.f;
#pragma unroll
    for (int r = 0; r < 8; ++r)
      C[(size_t)(m0 + r) * ldc + n] = acc[i][r] + bs;
  }
}

template <int KB>
static void launch_gemm(const __bf16* Ap, const __bf16* Bp, const float* bias,
                        float* C, int N, long ldc, hipStream_t stream) {
  int nt = N >> 4;                   // 16-wide tiles; all N are multiples of 64
  int waves = (nt + NT_ - 1) / NT_;  // 4 tiles per wave
  gemm16<KB><<<(waves + 3) / 4, 128, 0, stream>>>(Ap, Bp, bias, C, nt, ldc);
}

// LayerNorm over N per row (16 rows); optional SiLU; writes fp32 and/or
// packed A-fragments (fused pack epilogue for the next GEMM's A operand).
__global__ void ln_kernel(const float* __restrict__ X, int N,
                          const float* __restrict__ sc, const float* __restrict__ bi,
                          float* __restrict__ outF, __bf16* __restrict__ outP,
                          int dosilu) {
  int row = blockIdx.x;
  const float* x = X + (size_t)row * N;
  __shared__ float s1[256], s2[256];
  float a = 0.f, b = 0.f;
  for (int k = threadIdx.x; k < N; k += blockDim.x) {
    float v = x[k];
    a += v; b += v * v;
  }
  s1[threadIdx.x] = a; s2[threadIdx.x] = b;
  __syncthreads();
  for (int off = 128; off > 0; off >>= 1) {
    if ((int)threadIdx.x < off) {
      s1[threadIdx.x] += s1[threadIdx.x + off];
      s2[threadIdx.x] += s2[threadIdx.x + off];
    }
    __syncthreads();
  }
  float mu = s1[0] / N;
  float var = s2[0] / N - mu * mu;
  float rstd = rsqrtf(var + 1e-5f);
  for (int k = threadIdx.x; k < N; k += blockDim.x) {
    float v = (x[k] - mu) * rstd * sc[k] + bi[k];
    if (dosilu) v = v * sigm_(v);
    if (outF) outF[(size_t)row * N + k] = v;
    if (outP) outP[a_frag_off(row, k)] = (__bf16)v;
  }
}

// GRU gate math on LN'd parts [16 x 12288]; updates deter state, writes
// deter_new into d_out slice, packs deter_new into A5 (prior) and A7 (post).
__global__ void gru_kernel(const float* __restrict__ pln, float* __restrict__ deter,
                           float* __restrict__ out, __bf16* __restrict__ A5,
                           __bf16* __restrict__ A7, int t) {
  int i = blockIdx.x * blockDim.x + threadIdx.x;
  if (i >= 16 * DETER_) return;
  int m = i >> 12, j = i & (DETER_ - 1);
  const float* row = pln + (size_t)m * (3 * DETER_);
  float r = sigm_(row[j]);
  float c = tanhf(r * row[DETER_ + j]);
  float u = sigm_(row[2 * DETER_ + j] - 1.f);
  float dn = u * c + (1.f - u) * deter[i];
  deter[i] = dn;
  out[(size_t)m * T_ * OUTD_ + (size_t)t * OUTD_ + j] = dn;
  __bf16 v = (__bf16)dn;
  size_t fo = a_frag_off(m, j);
  A5[fo] = v;
  A7[fo] = v;
}

__global__ void init_states(float* __restrict__ deter, float* __restrict__ stoch,
                            const float* __restrict__ initd, const float* __restrict__ inits) {
  int i = blockIdx.x * blockDim.x + threadIdx.x;
  if (i < 16 * DETER_) deter[i] = initd[i & (DETER_ - 1)];
  if (i < 16 * STOCH_) stoch[i] = inits[i & (STOCH_ - 1)];
}

__global__ void mask_states(float* __restrict__ deter, float* __restrict__ stoch,
                            const float* __restrict__ initd, const float* __restrict__ inits,
                            const unsigned char* __restrict__ is_first, int t) {
  int i = blockIdx.x * blockDim.x + threadIdx.x;
  if (i >= 16 * DETER_) return;
  int b = i >> 12;
  if (is_first[b * T_ + t]) {
    deter[i] = initd[i & (DETER_ - 1)];
    if ((i & (DETER_ - 1)) < STOCH_)
      stoch[(b << 10) + (i & (STOCH_ - 1))] = inits[i & (STOCH_ - 1)];
  }
}

// softmax + unimix + deterministic hash-Gumbel categorical per (b,s);
// writes straight-through sample to d_out and the recurrent stoch state.
__global__ void sample_kernel(float* __restrict__ out, float* __restrict__ stoch,
                              int t) {
  int i = blockIdx.x * blockDim.x + threadIdx.x;
  if (i >= B_ * S_) return;
  int b = i / S_, s = i % S_;
  size_t lbase = (size_t)b * T_ * OUTD_ + (size_t)t * OUTD_ + 6144 + s * 32;
  float p[32];
  float mx = -1e30f;
#pragma unroll
  for (int c = 0; c < 32; ++c) { p[c] = out[lbase + c]; mx = fmaxf(mx, p[c]); }
  float se = 0.f;
#pragma unroll
  for (int c = 0; c < 32; ++c) { p[c] = __expf(p[c] - mx); se += p[c]; }
  float inv = 1.f / se;
  int best = 0; float bestv = -1e30f;
#pragma unroll
  for (int c = 0; c < 32; ++c) {
    float pr = 0.99f * p[c] * inv + 0.01f / 32.f;
    p[c] = pr;
    unsigned h = hash_u32(hash_u32(0x243F6A88u ^ (unsigned)(t * 0x9E3779B9u)) ^
                          (unsigned)(i * 0x85ebca6bu) ^ (unsigned)(c * 0x27d4eb2fu));
    float u = ((h >> 8) + 0.5f) * (1.f / 16777216.f);
    float g = -__logf(-__logf(u));
    float score = __logf(pr) + g;
    if (score > bestv) { bestv = score; best = c; }
  }
  size_t obase = (size_t)b * T_ * OUTD_ + (size_t)t * OUTD_ + DETER_ + s * 32;
#pragma unroll
  for (int c = 0; c < 32; ++c) {
    float oh = (c == best) ? 1.f : 0.f;
    float st = (oh + p[c]) - p[c];  // mimic one_hot + probs - stop_grad(probs)
    out[obase + c] = st;
    stoch[(size_t)b * STOCH_ + s * 32 + c] = st;
  }
}

// ---------------------------------------------------------------------------

extern "C" void kernel_launch(void* const* d_in, const int* in_sizes, int n_in,
                              void* d_out, int out_size, void* d_ws, size_t ws_size,
                              hipStream_t stream) {
  (void)in_sizes; (void)n_in; (void)out_size; (void)ws_size;
  const float* embed    = (const float*)d_in[0];
  const float* action   = (const float*)d_in[1];
  const unsigned char* is_first = (const unsigned char*)d_in[2];
  const float* W_act  = (const float*)d_in[3];
  const float* b_act  = (const float*)d_in[4];
  const float* W_st   = (const float*)d_in[5];
  const float* b_st   = (const float*)d_in[6];
  const float* W_gi   = (const float*)d_in[7];
  const float* b_gi   = (const float*)d_in[8];
  const float* gi_s   = (const float*)d_in[9];
  const float* gi_b   = (const float*)d_in[10];
  const float* W_gru  = (const float*)d_in[11];
  const float* gru_s  = (const float*)d_in[12];
  const float* gru_b  = (const float*)d_in[13];
  const float* W_p1   = (const float*)d_in[14];
  const float* b_p1   = (const float*)d_in[15];
  const float* p_s    = (const float*)d_in[16];
  const float* p_b    = (const float*)d_in[17];
  const float* W_p2   = (const float*)d_in[18];
  const float* b_p2   = (const float*)d_in[19];
  const float* W_q1   = (const float*)d_in[20];
  const float* b_q1   = (const float*)d_in[21];
  const float* q_s    = (const float*)d_in[22];
  const float* q_b    = (const float*)d_in[23];
  const float* W_q2   = (const float*)d_in[24];
  const float* b_q2   = (const float*)d_in[25];
  const float* init_d = (const float*)d_in[26];
  const float* init_s = (const float*)d_in[27];
  float* out = (float*)d_out;

  // ---- workspace carve (bump allocator, 256B aligned) ----
  char* ws = (char*)d_ws;
  size_t off = 0;
  auto alloc = [&](size_t bytes) -> char* {
    char* p = ws + off;
    off += (bytes + 255) & ~(size_t)255;
    return p;
  };
  // packed bf16 weights: [ntile][ktile][512] fragments
  __bf16* Wact_p = (__bf16*)alloc((size_t)1   * 64  * 512 * 2);
  __bf16* Wst_p  = (__bf16*)alloc((size_t)32  * 64  * 512 * 2);
  __bf16* Wgi_p  = (__bf16*)alloc((size_t)64  * 64  * 512 * 2);
  __bf16* Wgru_p = (__bf16*)alloc((size_t)160 * 768 * 512 * 2);
  __bf16* Wp1_p  = (__bf16*)alloc((size_t)128 * 64  * 512 * 2);
  __bf16* Wp2_p  = (__bf16*)alloc((size_t)32  * 64  * 512 * 2);
  __bf16* Wq1_p  = (__bf16*)alloc((size_t)160 * 64  * 512 * 2);
  __bf16* Wq2_p  = (__bf16*)alloc((size_t)32  * 64  * 512 * 2);
  // packed bf16 activation fragments
  __bf16* A_a    = (__bf16*)alloc((size_t)1   * 512 * 2);
  __bf16* A_st   = (__bf16*)alloc((size_t)32  * 512 * 2);
  __bf16* A_feat = (__bf16*)alloc((size_t)64  * 512 * 2);
  __bf16* A4     = (__bf16*)alloc((size_t)160 * 512 * 2);  // [x | deter]
  __bf16* A5     = (__bf16*)alloc((size_t)128 * 512 * 2);  // deter_new
  __bf16* A6     = (__bf16*)alloc((size_t)32  * 512 * 2);
  __bf16* A7     = (__bf16*)alloc((size_t)160 * 512 * 2);  // [deter_new | embed]
  __bf16* A8     = (__bf16*)alloc((size_t)32  * 512 * 2);
  // fp32 activations + states
  float* featb   = (float*)alloc((size_t)16 * 2048  * 4);
  float* gib     = (float*)alloc((size_t)16 * 1024  * 4);
  float* parts   = (float*)alloc((size_t)16 * 12288 * 4);
  float* partsLN = (float*)alloc((size_t)16 * 12288 * 4);
  float* p1b     = (float*)alloc((size_t)16 * 1024  * 4);
  float* q1b     = (float*)alloc((size_t)16 * 1024  * 4);
  float* deter   = (float*)alloc((size_t)16 * DETER_ * 4);
  float* stoch   = (float*)alloc((size_t)16 * STOCH_ * 4);

  auto pb = [&](const float* W, int K, int N, int Kb, __bf16* dst) {
    size_t total = (size_t)(N >> 4) * Kb * 512;
    pack_b<<<(unsigned)((total + 255) / 256), 256, 0, stream>>>(W, K, N, Kb, dst);
  };
  auto ln = [&](const float* X, int N, const float* sc, const float* bi,
                float* oF, __bf16* oP, int silu) {
    ln_kernel<<<16, 256, 0, stream>>>(X, N, sc, bi, oF, oP, silu);
  };
  auto pa = [&](const float* src, long ld, int Ksrc, int Kpad, __bf16* dst) {
    pack_a<<<(16 * Kpad + 255) / 256, 256, 0, stream>>>(src, ld, Ksrc, Kpad, dst);
  };

  // weights -> bf16 WMMA fragments (each call; no cross-call state allowed)
  pb(W_act, ACT_, H_,        1,   Wact_p);
  pb(W_st,  STOCH_, H_,      32,  Wst_p);
  pb(W_gi,  2 * H_, H_,      64,  Wgi_p);
  pb(W_gru, H_ + DETER_, 3 * DETER_, 160, Wgru_p);
  pb(W_p1,  DETER_, H_,      128, Wp1_p);
  pb(W_p2,  H_, STOCH_,      32,  Wp2_p);
  pb(W_q1,  DETER_ + E_, H_, 160, Wq1_p);
  pb(W_q2,  H_, STOCH_,      32,  Wq2_p);

  init_states<<<(16 * DETER_ + 255) / 256, 256, 0, stream>>>(deter, stoch, init_d, init_s);

  const long OLD = (long)T_ * OUTD_;  // d_out leading dim per batch row
  for (int t = 0; t < T_; ++t) {
    mask_states<<<(16 * DETER_ + 255) / 256, 256, 0, stream>>>(
        deter, stoch, init_d, init_s, is_first, t);
    // feat = [a @ W_act + b_act , stoch @ W_stoch + b_stoch]
    pa(action + (size_t)t * ACT_, (long)T_ * ACT_, ACT_, 32, A_a);
    pa(stoch, STOCH_, STOCH_, STOCH_, A_st);
    launch_gemm<1>(A_a,  Wact_p, b_act, featb,        H_, 2048, stream);
    launch_gemm<32>(A_st, Wst_p, b_st,  featb + 1024, H_, 2048, stream);
    // x = silu(LN(feat @ W_gi + b_gi)) -> packed into A4[k<1024]
    pa(featb, 2048, 2048, 2048, A_feat);
    launch_gemm<64>(A_feat, Wgi_p, b_gi, gib, H_, H_, stream);
    ln(gib, H_, gi_s, gi_b, nullptr, A4, 1);
    pa(deter, DETER_, DETER_, DETER_, A4 + (size_t)32 * 512);  // A4[k>=1024] = deter
    // parts = LN(concat(x, deter) @ W_gru)
    launch_gemm<160>(A4, Wgru_p, nullptr, parts, 3 * DETER_, 3 * DETER_, stream);
    ln(parts, 3 * DETER_, gru_s, gru_b, partsLN, nullptr, 0);
    gru_kernel<<<(16 * DETER_ + 255) / 256, 256, 0, stream>>>(
        partsLN, deter, out, A5, A7, t);
    pa(embed + (size_t)t * E_, (long)T_ * E_, E_, E_, A7 + (size_t)128 * 512);
    // prior logits -> out[..., 5120:6144]  (prior sample is dead code in ref)
    launch_gemm<128>(A5, Wp1_p, b_p1, p1b, H_, H_, stream);
    ln(p1b, H_, p_s, p_b, nullptr, A6, 1);
    launch_gemm<32>(A6, Wp2_p, b_p2, out + (size_t)t * OUTD_ + 5120, STOCH_, OLD, stream);
    // post logits -> out[..., 6144:7168]
    launch_gemm<160>(A7, Wq1_p, b_q1, q1b, H_, H_, stream);
    ln(q1b, H_, q_s, q_b, nullptr, A8, 1);
    launch_gemm<32>(A8, Wq2_p, b_q2, out + (size_t)t * OUTD_ + 6144, STOCH_, OLD, stream);
    // post sample -> out[..., 4096:5120] and recurrent stoch state
    sample_kernel<<<(B_ * S_ + 255) / 256, 256, 0, stream>>>(out, stoch, t);
  }
}
